// SlotAttention_2525440770085
// MI455X (gfx1250) — matrix operand order repaired and checked by reading, not compiled, not run on hardware
//
#include <hip/hip_runtime.h>
#include <hip/hip_bf16.h>
#include <math.h>

typedef __attribute__((ext_vector_type(16))) _Float16 v16h;
typedef __attribute__((ext_vector_type(8)))  _Float16 v8h;
typedef __attribute__((ext_vector_type(8)))  float    v8f;

#define B_   32
#define N_   16384
#define DF_  128
#define DS_  64
#define NS_  8
#define HID_ 128
#define EPS_LN   1e-3f
#define EPS_ATTN 1e-8f
#define SCALE_   0.08838834764831845f   // DF^-0.5

#define USE_ASYNC_LDS 1   // global_load_async_to_lds_b128 weight staging (ASYNCcnt path)

static __device__ __forceinline__ v8f wmma16(v16h a, v16h b, v8f c) {
  // D = A(16x32 f16) * B(32x16 f16) + C(16x16 f32)
  return __builtin_amdgcn_wmma_f32_16x16x32_f16(false, a, false, b, (short)0, c, false, false);
}

static __device__ __forceinline__ float sigmoidf_(float x) { return 1.f / (1.f + __expf(-x)); }

#if USE_ASYNC_LDS
// LDS byte-offset of a generic pointer that points into shared memory.
static __device__ __forceinline__ unsigned lds_off(const void* p) {
  return (unsigned)(size_t)(__attribute__((address_space(3))) const char*)p;
}
static __device__ __forceinline__ void async_copy_b128(unsigned lds_byte_off, const void* gsrc) {
  asm volatile("global_load_async_to_lds_b128 %0, %1, off"
               :: "v"(lds_byte_off), "v"(gsrc) : "memory");
}
static __device__ __forceinline__ void wait_asynccnt0() {
  asm volatile("s_wait_asynccnt 0" ::: "memory");
}
#endif

// ---------------- weight fp32 -> fp16 conversion (once) ----------------
__global__ __launch_bounds__(256) void wconv_kernel(const float* __restrict__ Wk,
                                                    const float* __restrict__ Wv,
                                                    _Float16* __restrict__ Wk_h,
                                                    _Float16* __restrict__ Wv_h) {
  int i = blockIdx.x * 256 + threadIdx.x;
  if (i < DS_ * DF_) { Wk_h[i] = (_Float16)Wk[i]; Wv_h[i] = (_Float16)Wv[i]; }
}

// ---------------- fused LayerNorm + K/V projection (WMMA) ----------------
// Each block: 128 tokens (8 waves x 16 tokens). Per wave: A = LN(x) tile 16x128 f16,
// B = weight rows from LDS, 32 WMMAs -> k (tok-major f16), v (feature-major f16).
__global__ __launch_bounds__(256) void kv_kernel(
    const float* __restrict__ x, const float* __restrict__ lng, const float* __restrict__ lnb,
    const _Float16* __restrict__ Wk_h, const float* __restrict__ bk,
    const _Float16* __restrict__ Wv_h, const float* __restrict__ bv,
    _Float16* __restrict__ k_out,     // (B, N, DS)
    _Float16* __restrict__ v_out) {   // (B, DS, N)
  __shared__ _Float16 sW[2][DS_ * DF_];   // 32 KB
  __shared__ float sg[DF_], sb[DF_];
  int tid = threadIdx.x;
#if USE_ASYNC_LDS
  {
    unsigned base = lds_off(&sW[0][0]);
    // 32 KB in 16B chunks: 2048 chunks, 8 async wave-ops per wave
    for (int i = tid; i < 2048; i += 256) {
      const void* src = (i < 1024) ? (const void*)(Wk_h + (size_t)i * 8)
                                   : (const void*)(Wv_h + (size_t)(i - 1024) * 8);
      async_copy_b128(base + (unsigned)i * 16u, src);
    }
    wait_asynccnt0();
  }
#else
  for (int i = tid; i < DS_ * DF_; i += 256) { sW[0][i] = Wk_h[i]; sW[1][i] = Wv_h[i]; }
#endif
  if (tid < DF_) { sg[tid] = lng[tid]; sb[tid] = lnb[tid]; }
  __syncthreads();

  int wave = tid >> 5, lane = tid & 31;
  int r = lane & 15, h = lane >> 4;
  size_t row0 = (size_t)blockIdx.x * 128 + (size_t)wave * 16;  // b*N + tok
  int b    = (int)(row0 / N_);
  int tok0 = (int)(row0 - (size_t)b * N_);
  const float* xrow = x + (row0 + (size_t)r) * DF_;

  // lane (r,h) owns row r, K in {16c + 8h + j : c=0..7, j=0..7}  (A-frag pattern)
  float xr[64];
  float sum = 0.f, ssq = 0.f;
#pragma unroll
  for (int c = 0; c < 8; ++c)
#pragma unroll
    for (int j = 0; j < 8; ++j) {
      float t = xrow[16 * c + h * 8 + j];
      xr[c * 8 + j] = t; sum += t; ssq += t * t;
    }
  sum += __shfl_xor(sum, 16, 32);
  ssq += __shfl_xor(ssq, 16, 32);
  float mean = sum * (1.f / DF_);
  float rstd = rsqrtf(ssq * (1.f / DF_) - mean * mean + EPS_LN);

  v16h afrag[4];
#pragma unroll
  for (int c = 0; c < 8; ++c)
#pragma unroll
    for (int j = 0; j < 8; ++j) {
      int K = 16 * c + h * 8 + j;
      float t = (xr[c * 8 + j] - mean) * rstd * sg[K] + sb[K];
      afrag[c >> 1][(c & 1) * 8 + j] = (_Float16)t;
    }

#pragma unroll
  for (int m = 0; m < 2; ++m) {
    const _Float16* W = sW[m];
    const float* bias = m ? bv : bk;
#pragma unroll
    for (int nt = 0; nt < 4; ++nt) {
      float bn = bias[nt * 16 + r];
      v8f c = { bn, bn, bn, bn, bn, bn, bn, bn };
#pragma unroll
      for (int kb = 0; kb < 4; ++kb) {
        const _Float16* wp = &W[(nt * 16 + r) * DF_ + kb * 32 + h * 8];  // two 16B halves: k 8h..8h+7, 16+8h..16+8h+7
        v8h blo = *(const v8h*)wp;
        v8h bhi = *(const v8h*)(wp + 16);
        v16h bf = __builtin_shufflevector(blo, bhi, 0, 1, 2, 3, 4, 5, 6, 7, 8, 9, 10, 11, 12, 13, 14, 15);
        c = wmma16(afrag[kb], bf, c);
      }
      if (m == 0) {
        // k tok-major: lane holds feature nt*16+r for tokens h*8+g (32B-contig per half-wave)
#pragma unroll
        for (int g = 0; g < 8; ++g)
          k_out[(row0 + (size_t)(h * 8 + g)) * DS_ + nt * 16 + r] = (_Float16)c[g];
      } else {
        // v feature-major: lane's 8 values are contiguous tokens -> one 16B store
        v8h pk;
#pragma unroll
        for (int g = 0; g < 8; ++g) pk[g] = (_Float16)c[g];
        *(v8h*)&v_out[((size_t)b * DS_ + nt * 16 + r) * N_ + tok0 + h * 8] = pk;
      }
    }
  }
}

// ---------------- per-iteration slot prep: q = LN(slots)@Wq^T + bq ; zero U,S ----------------
__global__ __launch_bounds__(64) void prep_kernel(
    const float* __restrict__ slots, const float* __restrict__ lng, const float* __restrict__ lnb,
    const float* __restrict__ Wq, const float* __restrict__ bq,
    _Float16* __restrict__ q_h, float* __restrict__ U, float* __restrict__ S) {
  int b = blockIdx.x, t = threadIdx.x;
  __shared__ float sh[NS_][DS_];
  __shared__ float mu[NS_], rs[NS_];
  if (t < NS_) {
    const float* row = slots + (b * NS_ + t) * DS_;
    float sum = 0.f, ssq = 0.f;
    for (int k = 0; k < DS_; ++k) { float v = row[k]; sum += v; ssq += v * v; }
    float m = sum / DS_;
    mu[t] = m; rs[t] = rsqrtf(ssq / DS_ - m * m + EPS_LN);
  }
  __syncthreads();
  for (int s = 0; s < NS_; ++s)
    sh[s][t] = (slots[(b * NS_ + s) * DS_ + t] - mu[s]) * rs[s] * lng[t] + lnb[t];
  __syncthreads();
  for (int s = 0; s < NS_; ++s) {
    float acc = bq[t];
    for (int k = 0; k < DS_; ++k) acc += sh[s][k] * Wq[t * DS_ + k];
    q_h[(b * NS_ + s) * DS_ + t] = (_Float16)acc;
  }
  for (int i = t; i < NS_ * DS_; i += 64) U[b * NS_ * DS_ + i] = 0.f;
  if (t < NS_) S[b * NS_ + t] = 0.f;
}

// ---------------- attention phase: dots (WMMA) -> softmax over slots (lane-local)
//                  -> U += attn @ v (WMMA), S += row-sums. grid = B x 16 chunks.
__global__ __launch_bounds__(256) void attn_kernel(
    const _Float16* __restrict__ k_h, const _Float16* __restrict__ v_t,
    const _Float16* __restrict__ q_h,
    float* __restrict__ U, float* __restrict__ S) {
  __shared__ _Float16 sattn[8][16][32];  // per-wave attn staging (C->A transpose)
  __shared__ float redU[NS_][DS_];
  __shared__ float redS[NS_];
  int tid = threadIdx.x;
  int b = blockIdx.x >> 4, cb = blockIdx.x & 15;
  int wave = tid >> 5, lane = tid & 31;
  int r = lane & 15, h = lane >> 4;

  for (int i = tid; i < NS_ * DS_; i += 256) redU[i >> 6][i & 63] = 0.f;
  if (tid < NS_) redS[tid] = 0.f;
  __syncthreads();

  // q A-fragments (M rows 8..15 are zero padding)
  v16h aq0, aq1;
#pragma unroll
  for (int g = 0; g < 8; ++g)
#pragma unroll
    for (int e = 0; e < 2; ++e) {
      int K = h * 16 + g * 2 + e;
      _Float16 q0 = (_Float16)0.f, q1 = (_Float16)0.f;
      if (r < NS_) {
        q0 = q_h[(b * NS_ + r) * DS_ + K];
        q1 = q_h[(b * NS_ + r) * DS_ + 32 + K];
      }
      aq0[2 * g + e] = q0; aq1[2 * g + e] = q1;
    }

  v8f cu[4] = {};
  float sacc[8];
#pragma unroll
  for (int g = 0; g < 8; ++g) sacc[g] = 0.f;

  for (int ch = 0; ch < 4; ++ch) {        // 4 chunks of 32 tokens per wave
    int tbase = cb * 1024 + wave * 128 + ch * 32;
    v8f cd[2];
#pragma unroll
    for (int t2 = 0; t2 < 2; ++t2) {
      int tok = tbase + t2 * 16 + r;
      const _Float16* kp = k_h + ((size_t)b * N_ + tok) * DS_ + h * 16;
      v16h bk0 = *(const v16h*)kp;
      v16h bk1 = *(const v16h*)(kp + 32);
      __builtin_prefetch(kp + 32 * DS_, 0, 0);   // next chunk -> global_prefetch_b8
      v8f c = {};
      c = wmma16(aq0, bk0, c);
      c = wmma16(aq1, bk1, c);
      cd[t2] = c;
    }
    // softmax over the 8 slots of this lane's token column (pure VALU)
    float at[2][8];
#pragma unroll
    for (int t2 = 0; t2 < 2; ++t2) {
      float mx = -3.0e38f;
#pragma unroll
      for (int g = 0; g < 8; ++g) { float d = cd[t2][g] * SCALE_; at[t2][g] = d; mx = fmaxf(mx, d); }
      float sum = 0.f;
#pragma unroll
      for (int g = 0; g < 8; ++g) { float e = __expf(at[t2][g] - mx); at[t2][g] = e; sum += e; }
      float inv = 1.f / sum;
#pragma unroll
      for (int g = 0; g < 8; ++g) {
        float a = at[t2][g] * inv + EPS_ATTN;
        at[t2][g] = a;
        sacc[g] += a;                 // lanes 16..31 accumulate garbage, filtered at flush
      }
    }
    // C-layout -> A-layout transpose via LDS (wave-private; DS ops are in-order)
#pragma unroll
    for (int g = 0; g < 8; ++g) {
      sattn[wave][h * 8 + g][r]      = (_Float16)at[0][g];
      sattn[wave][h * 8 + g][16 + r] = (_Float16)at[1][g];
    }
    v16h aat;
#pragma unroll
    for (int g = 0; g < 8; ++g) {
      int K = h * 16 + g * 2;
      aat[2 * g]     = sattn[wave][r][K];
      aat[2 * g + 1] = sattn[wave][r][K + 1];
    }
    // U accumulation: B columns = features, K = this chunk's 32 tokens (contiguous in v_t)
#pragma unroll
    for (int dt = 0; dt < 4; ++dt) {
      const _Float16* vp = v_t + ((size_t)b * DS_ + dt * 16 + r) * N_ + tbase + h * 16;
      v16h bvf = *(const v16h*)vp;
      __builtin_prefetch(vp + 32, 0, 0);
      cu[dt] = wmma16(aat, bvf, cu[dt]);
    }
  }

  if (h == 0) {  // slot rows 0..7 live in lanes 0..15
#pragma unroll
    for (int dt = 0; dt < 4; ++dt)
#pragma unroll
      for (int g = 0; g < 8; ++g)
        atomicAdd(&redU[g][dt * 16 + r], cu[dt][g]);
#pragma unroll
    for (int g = 0; g < 8; ++g) atomicAdd(&redS[g], sacc[g]);
  }
  __syncthreads();
  for (int i = tid; i < NS_ * DS_; i += 256) atomicAdd(&U[b * NS_ * DS_ + i], redU[i >> 6][i & 63]);
  if (tid < NS_) atomicAdd(&S[b * NS_ + tid], redS[tid]);
}

// ---------------- GRU + LN + MLP (tiny, scalar f32, one block per batch) ----------------
__global__ __launch_bounds__(256) void gru_mlp_kernel(
    const float* __restrict__ slots_in, const float* __restrict__ U, const float* __restrict__ S,
    const float* __restrict__ W_ih, const float* __restrict__ b_ih,
    const float* __restrict__ W_hh, const float* __restrict__ b_hh,
    const float* __restrict__ lng, const float* __restrict__ lnb,
    const float* __restrict__ W1, const float* __restrict__ b1,
    const float* __restrict__ W2, const float* __restrict__ b2,
    float* __restrict__ slots_out, float* __restrict__ dout) {
  int b = blockIdx.x, t = threadIdx.x;
  __shared__ float upd[NS_][DS_], hh[NS_][DS_], gi[NS_][3 * DS_], gh[NS_][3 * DS_];
  __shared__ float hn[NS_][DS_], mm[NS_][DS_], hid[NS_][HID_];
  __shared__ float mu[NS_], rs[NS_];

  for (int i = t; i < NS_ * DS_; i += 256) {
    int s = i >> 6, d = i & 63;
    upd[s][d] = U[b * NS_ * DS_ + i] / S[b * NS_ + s];   // fold attn row-normalization
    hh[s][d] = slots_in[b * NS_ * DS_ + i];
  }
  __syncthreads();
  for (int i = t; i < NS_ * 3 * DS_; i += 256) {
    int s = i / (3 * DS_), j = i % (3 * DS_);
    float a = b_ih[j], c = b_hh[j];
    for (int k = 0; k < DS_; ++k) { a += upd[s][k] * W_ih[j * DS_ + k]; c += hh[s][k] * W_hh[j * DS_ + k]; }
    gi[s][j] = a; gh[s][j] = c;
  }
  __syncthreads();
  for (int i = t; i < NS_ * DS_; i += 256) {
    int s = i >> 6, d = i & 63;
    float rr = sigmoidf_(gi[s][d] + gh[s][d]);
    float zz = sigmoidf_(gi[s][DS_ + d] + gh[s][DS_ + d]);
    float nn = tanhf(gi[s][2 * DS_ + d] + rr * gh[s][2 * DS_ + d]);
    hn[s][d] = (1.f - zz) * nn + zz * hh[s][d];
  }
  __syncthreads();
  if (t < NS_) {
    float sum = 0.f, ssq = 0.f;
    for (int k = 0; k < DS_; ++k) { float v = hn[t][k]; sum += v; ssq += v * v; }
    float m = sum / DS_;
    mu[t] = m; rs[t] = rsqrtf(ssq / DS_ - m * m + EPS_LN);
  }
  __syncthreads();
  for (int i = t; i < NS_ * DS_; i += 256) {
    int s = i >> 6, d = i & 63;
    mm[s][d] = (hn[s][d] - mu[s]) * rs[s] * lng[d] + lnb[d];
  }
  __syncthreads();
  for (int i = t; i < NS_ * HID_; i += 256) {
    int s = i >> 7, j = i & 127;
    float a = b1[j];
    for (int k = 0; k < DS_; ++k) a += mm[s][k] * W1[j * DS_ + k];
    hid[s][j] = fmaxf(a, 0.f);
  }
  __syncthreads();
  for (int i = t; i < NS_ * DS_; i += 256) {
    int s = i >> 6, d = i & 63;
    float a = b2[d];
    for (int k = 0; k < HID_; ++k) a += hid[s][k] * W2[d * HID_ + k];
    float o = hn[s][d] + a;
    slots_out[b * NS_ * DS_ + i] = o;
    dout[b * NS_ * DS_ + i] = o;
  }
}

extern "C" void kernel_launch(void* const* d_in, const int* in_sizes, int n_in,
                              void* d_out, int out_size, void* d_ws, size_t ws_size,
                              hipStream_t stream) {
  (void)in_sizes; (void)n_in; (void)out_size; (void)ws_size;
  const float* inputs  = (const float*)d_in[0];
  const float* slots0  = (const float*)d_in[1];
  const float* ln_in_g = (const float*)d_in[2];
  const float* ln_in_b = (const float*)d_in[3];
  const float* ln_sl_g = (const float*)d_in[4];
  const float* ln_sl_b = (const float*)d_in[5];
  const float* ln_mlp_g= (const float*)d_in[6];
  const float* ln_mlp_b= (const float*)d_in[7];
  const float* Wq  = (const float*)d_in[8];
  const float* bq  = (const float*)d_in[9];
  const float* Wk  = (const float*)d_in[10];
  const float* bk  = (const float*)d_in[11];
  const float* Wv  = (const float*)d_in[12];
  const float* bv  = (const float*)d_in[13];
  const float* W_ih= (const float*)d_in[14];
  const float* b_ih= (const float*)d_in[15];
  const float* W_hh= (const float*)d_in[16];
  const float* b_hh= (const float*)d_in[17];
  const float* W1  = (const float*)d_in[18];
  const float* b1  = (const float*)d_in[19];
  const float* W2  = (const float*)d_in[20];
  const float* b2  = (const float*)d_in[21];

  char* ws = (char*)d_ws;
  size_t off = 0;
  auto alloc = [&](size_t bytes) -> void* {
    void* p = ws + off;
    off = (off + bytes + 255) & ~(size_t)255;
    return p;
  };
  _Float16* Wk_h   = (_Float16*)alloc((size_t)DS_ * DF_ * 2);
  _Float16* Wv_h   = (_Float16*)alloc((size_t)DS_ * DF_ * 2);
  _Float16* q_h    = (_Float16*)alloc((size_t)B_ * NS_ * DS_ * 2);
  float*    U      = (float*)   alloc((size_t)B_ * NS_ * DS_ * 4);
  float*    S      = (float*)   alloc((size_t)B_ * NS_ * 4);
  float*    slotsw = (float*)   alloc((size_t)B_ * NS_ * DS_ * 4);
  _Float16* k_h    = (_Float16*)alloc((size_t)B_ * N_ * DS_ * 2);   // 64 MB
  _Float16* v_t    = (_Float16*)alloc((size_t)B_ * N_ * DS_ * 2);   // 64 MB

  wconv_kernel<<<(DS_ * DF_ + 255) / 256, 256, 0, stream>>>(Wk, Wv, Wk_h, Wv_h);
  kv_kernel<<<(B_ * N_) / 128, 256, 0, stream>>>(inputs, ln_in_g, ln_in_b,
                                                 Wk_h, bk, Wv_h, bv, k_h, v_t);
  const float* cur = slots0;
  for (int it = 0; it < 2; ++it) {
    prep_kernel<<<B_, 64, 0, stream>>>(cur, ln_sl_g, ln_sl_b, Wq, bq, q_h, U, S);
    attn_kernel<<<B_ * 16, 256, 0, stream>>>(k_h, v_t, q_h, U, S);
    gru_mlp_kernel<<<B_, 256, 0, stream>>>(cur, U, S, W_ih, b_ih, W_hh, b_hh,
                                           ln_mlp_g, ln_mlp_b, W1, b1, W2, b2,
                                           slotsw, (float*)d_out);
    cur = slotsw;
  }
}
